// MMPNTimeFree_57647051047688
// MI455X (gfx1250) — matrix operand order repaired and checked by reading, not compiled, not run on hardware
//
#include <hip/hip_runtime.h>
#include <hip/hip_bf16.h>
#include <stdint.h>

// ---------------------------------------------------------------------------
// Problem constants (match reference)
// ---------------------------------------------------------------------------
#define NG    64
#define NPG   256
#define EPG   2048
#define NNODE (NG * NPG)    // 16384
#define NEDGE (NG * EPG)    // 131072
#define FN    256
#define FE    64
#define FG    128

typedef __attribute__((ext_vector_type(16))) __bf16 v16bf;
typedef __attribute__((ext_vector_type(8)))  float  v8f;
typedef __attribute__((ext_vector_type(4))) unsigned int u32x4;
typedef __attribute__((ext_vector_type(8))) int          i32x8;
typedef __attribute__((ext_vector_type(4))) int          i32x4;

#if defined(__has_builtin)
#  if __has_builtin(__builtin_amdgcn_tensor_load_to_lds) && \
      __has_builtin(__builtin_amdgcn_s_wait_tensorcnt)
#    define HAVE_TDM 1
#  endif
#endif
#ifndef HAVE_TDM
#  define HAVE_TDM 0
#endif

static __device__ __forceinline__ uint16_t f2bf(float f) {
  uint32_t u = __float_as_uint(f);
  uint32_t r = (u + 0x7FFFu + ((u >> 16) & 1u)) >> 16;  // round-to-nearest-even
  return (uint16_t)r;
}

// ---------------------------------------------------------------------------
// Small helper kernels
// ---------------------------------------------------------------------------

// Convert fp32 W[K][N] -> bf16 Wt[N][K] (transpose so GEMM B-tiles are
// contiguous in K, enabling b128/TDM row loads with no on-the-fly transpose).
__global__ void k_cvt_transpose_w(const float* __restrict__ w,
                                  uint16_t* __restrict__ wt, int K, int N) {
  int t = blockIdx.x * blockDim.x + threadIdx.x;   // over N*K, k fastest
  if (t >= K * N) return;
  int n = t / K;
  int k = t - n * K;
  wt[(size_t)n * K + k] = f2bf(w[(size_t)k * N + n]);
}

__global__ void k_zero_f32(float* __restrict__ p, int n) {
  int i = blockIdx.x * blockDim.x + threadIdx.x;
  if (i < n) p[i] = 0.0f;
}

// concat1[e][0:704] = [nodes[src[e]] | edge_attr[e] | nodes[tgt[e]] | glob[e/EPG]]
__global__ void k_concat_edges(const float* __restrict__ nodes,
                               const float* __restrict__ edge_attr,
                               const float* __restrict__ glob,
                               const int* __restrict__ src,
                               const int* __restrict__ tgt,
                               uint16_t* __restrict__ out) {
  int t = blockIdx.x * blockDim.x + threadIdx.x;   // NEDGE*704 < 2^31
  if (t >= NEDGE * 704) return;
  int e = t / 704;
  int k = t - e * 704;
  float v;
  if (k < 256)      v = nodes[(size_t)src[e] * FN + k];
  else if (k < 320) v = edge_attr[(size_t)e * FE + (k - 256)];
  else if (k < 576) v = nodes[(size_t)tgt[e] * FN + (k - 320)];
  else              v = glob[(size_t)(e >> 11) * FG + (k - 576)];
  out[t] = f2bf(v);
}

// concat2[i][0:640] = [nodes[i] | aggr[i] | glob[i/NPG]]
__global__ void k_concat_nodes(const float* __restrict__ nodes,
                               const float* __restrict__ aggr,
                               const float* __restrict__ glob,
                               uint16_t* __restrict__ out) {
  int t = blockIdx.x * blockDim.x + threadIdx.x;   // NNODE*640
  if (t >= NNODE * 640) return;
  int i = t / 640;
  int k = t - i * 640;
  float v;
  if (k < 256)      v = nodes[(size_t)i * FN + k];
  else if (k < 512) v = aggr[(size_t)i * 256 + (k - 256)];
  else              v = glob[(size_t)(i >> 8) * FG + (k - 512)];
  out[t] = f2bf(v);
}

// concatG[g][0:384] = [agg_nodes[g] | global_attr[g]]
__global__ void k_concat_glob(const float* __restrict__ aggN,
                              const float* __restrict__ glob,
                              uint16_t* __restrict__ out) {
  int t = blockIdx.x * blockDim.x + threadIdx.x;   // NG*384
  if (t >= NG * 384) return;
  int g = t / 384;
  int k = t - g * 384;
  float v = (k < 256) ? aggN[(size_t)g * 256 + k]
                      : glob[(size_t)g * FG + (k - 256)];
  out[t] = f2bf(v);
}

// action = softmax(G @ act_w + act_b) ; 64 rows x 8 cols (tiny)
__global__ void k_act_softmax(const float* __restrict__ G,
                              const float* __restrict__ w,
                              const float* __restrict__ b,
                              float* __restrict__ out) {
  int g = threadIdx.x;
  if (g >= NG) return;
  float lg[8];
  for (int j = 0; j < 8; ++j) lg[j] = b[j];
  for (int k = 0; k < 256; ++k) {
    float gv = G[g * 256 + k];
    for (int j = 0; j < 8; ++j) lg[j] += gv * w[k * 8 + j];
  }
  float mx = lg[0];
  for (int j = 1; j < 8; ++j) mx = fmaxf(mx, lg[j]);
  float s = 0.f;
  for (int j = 0; j < 8; ++j) { lg[j] = expf(lg[j] - mx); s += lg[j]; }
  float inv = 1.0f / s;
  for (int j = 0; j < 8; ++j) out[g * 8 + j] = lg[j] * inv;
}

// ---------------------------------------------------------------------------
// TDM tile load: DMA a rows x 32-elem bf16 tile (row stride = K elems) from
// global into LDS at lds_byte_off, with hardware row padding 64B->96B pitch.
// D# bitfields per CDNA5 ISA ch.8 (groups 0/1; groups 2/3 zero => 2D tensor).
// ---------------------------------------------------------------------------
#define TM  128
#define TK  32
#define LDT 48   // LDS row pitch (elements) -> 96B = 64B data + 32B pad

#if HAVE_TDM
static __device__ __forceinline__ void tdm_load_tile(uint32_t lds_byte_off,
                                                     const void* gptr,
                                                     int rows,
                                                     int rowStrideElems) {
  uint64_t ga = (uint64_t)(uintptr_t)gptr;
  u32x4 g0;
  g0[0] = 1u;                                    // count=1 (valid), user mode
  g0[1] = lds_byte_off;                          // lds_addr
  g0[2] = (uint32_t)ga;                          // global_addr[31:0]
  g0[3] = (uint32_t)((ga >> 32) & 0x01FFFFFFu)   // global_addr[56:32]
        | (2u << 30);                            // type=2
  i32x8 g1;
  g1[0] = (1 << 16)      // data_size = 2 bytes
        | (1 << 20)      // pad_enable
        | (3 << 22)      // pad_interval: 16 DWORDs (64B) between pads
        | (7 << 25);     // pad_amount: 8 DWORDs (32B) -> 96B LDS pitch
  const uint32_t td0 = 0x7FFFFFFFu;   // huge tensor dims: no OOB clipping
  const uint32_t td1 = 0x7FFFFFFFu;
  g1[1] = (int)((td0 & 0xFFFFu) << 16);                         // dim0[15:0]
  g1[2] = (int)(((td0 >> 16) & 0xFFFFu) | ((td1 & 0xFFFFu) << 16));
  g1[3] = (int)(((td1 >> 16) & 0xFFFFu) | ((uint32_t)TK << 16)); // tile_dim0=32
  g1[4] = rows & 0xFFFF;                                        // tile_dim1
  uint64_t st = (uint64_t)(uint32_t)rowStrideElems;             // dim0 stride
  g1[5] = (int)(uint32_t)st;
  g1[6] = (int)((st >> 32) & 0xFFFFu);
  g1[7] = 0;
  i32x4 z4; z4[0] = 0; z4[1] = 0; z4[2] = 0; z4[3] = 0;
  i32x8 z8;
#pragma unroll
  for (int i = 0; i < 8; ++i) z8[i] = 0;
  // amdgpu-toolchain (clang-23) 6-arg form:
  // (g0 u32x4, g1 i32x8, g2 i32x4, g3 i32x4, i32x8, cpol i32)
  __builtin_amdgcn_tensor_load_to_lds(g0, g1, z4, z4, z8, 0);
}
#endif

// ---------------------------------------------------------------------------
// bf16 WMMA GEMM:  C[M,N] = relu(A[M,K] @ Wt^T + bias)   (Wt is [N][K] bf16)
//   8 waves as 4(M) x 2(N); wave tile = 32 x (NF*16); block = 128 x (NF*32).
//   TDM path (MGUARD=false): double-buffered LDS, wave0 issues tensor DMA for
//   tile i+1 after the barrier, s_wait_tensorcnt publishes tile i -> one
//   barrier per K-step, DMA overlaps WMMA.
//   MGUARD path (glob GEMM, M=64): synchronous b128 vector loads with guards.
//   Epilogue (any combination): bf16 store, f32 store, scatter-atomicMax
//   (values >= 0 post-ReLU, so int-bit atomicMax over zero-init is exact max).
// ---------------------------------------------------------------------------
template <bool MGUARD, int NF>
__launch_bounds__(256)
__global__ void k_gemm_bf16(const uint16_t* __restrict__ A,
                            const uint16_t* __restrict__ Wt,
                            const float* __restrict__ bias,
                            uint16_t* __restrict__ outB,
                            float* __restrict__ outF,
                            float* __restrict__ scat,
                            const int* __restrict__ sidx,
                            int M, int N, int K) {
  constexpr int TN = NF * 32;               // 64 or 128
  __shared__ __align__(16) uint16_t As[2][TM][LDT];
  __shared__ __align__(16) uint16_t Bs[2][TN][LDT];   // Bs[buf][n][k]

  const int tid    = threadIdx.x;
  const int lane   = tid & 31;
  const int wid    = tid >> 5;
  const int waveM  = wid & 3;    // 0..3  -> 32-row strip
  const int waveN  = wid >> 2;   // 0..1  -> (NF*16)-col strip
  const int mBlock = blockIdx.y * TM;
  const int nBlock = blockIdx.x * TN;

  const int halfSel = lane >> 4;  // K-half select per ISA 16-bit A/B layouts
  const int l16     = lane & 15;

  v8f acc[2][NF] = {};
  union Frag { uint4 q[2]; v16bf v; };

  auto compute = [&](int buf) {
    Frag fa[2], fb[NF];
#pragma unroll
    for (int fm = 0; fm < 2; ++fm) {
      int m = waveM * 32 + fm * 16 + l16;
      fa[fm].q[0] = *(const uint4*)&As[buf][m][halfSel * 8];      // K 0..7/8..15
      fa[fm].q[1] = *(const uint4*)&As[buf][m][16 + halfSel * 8]; // K 16..23/24..31
    }
#pragma unroll
    for (int fn = 0; fn < NF; ++fn) {
      int n = waveN * (NF * 16) + fn * 16 + l16;
      fb[fn].q[0] = *(const uint4*)&Bs[buf][n][halfSel * 8];
      fb[fn].q[1] = *(const uint4*)&Bs[buf][n][16 + halfSel * 8];
    }
#pragma unroll
    for (int fm = 0; fm < 2; ++fm)
#pragma unroll
      for (int fn = 0; fn < NF; ++fn)
        acc[fm][fn] = __builtin_amdgcn_wmma_f32_16x16x32_bf16(
            false, fa[fm].v, false, fb[fn].v,
            (short)0, acc[fm][fn], false, false);
  };

#if HAVE_TDM
  if constexpr (!MGUARD) {
    const uint16_t* Abase = A + (size_t)mBlock * K;
    const uint16_t* Bbase = Wt + (size_t)nBlock * K;
    const int nsteps = K / TK;
    if (wid == 0) {
      tdm_load_tile((uint32_t)(uintptr_t)&As[0][0][0], Abase, TM, K);
      tdm_load_tile((uint32_t)(uintptr_t)&Bs[0][0][0], Bbase, TN, K);
    }
    for (int i = 0; i < nsteps; ++i) {
      if (wid == 0) __builtin_amdgcn_s_wait_tensorcnt(0);  // tile i landed
      __syncthreads();                                     // publish tile i;
                                                           // all done reading buf^1
      if (wid == 0 && i + 1 < nsteps) {                    // DMA tile i+1 || WMMA
        int nb = (i + 1) & 1;
        tdm_load_tile((uint32_t)(uintptr_t)&As[nb][0][0],
                      Abase + (size_t)(i + 1) * TK, TM, K);
        tdm_load_tile((uint32_t)(uintptr_t)&Bs[nb][0][0],
                      Bbase + (size_t)(i + 1) * TK, TN, K);
      }
      compute(i & 1);
    }
  } else
#endif
  {
    // synchronous vector-load path (MGUARD, or TDM builtin unavailable)
    const int c4 = tid & 3;         // uint4 column within 32-wide K tile
    const int rT = tid >> 2;        // 0..63
    for (int k0 = 0; k0 < K; k0 += TK) {
#pragma unroll
      for (int p = 0; p < 2; ++p) {
        int r = rT + p * 64;
        uint4 v;
        if (MGUARD) {
          v = make_uint4(0u, 0u, 0u, 0u);
          if (mBlock + r < M)
            v = *(const uint4*)(A + (size_t)(mBlock + r) * K + k0 + c4 * 8);
        } else {
          v = *(const uint4*)(A + (size_t)(mBlock + r) * K + k0 + c4 * 8);
        }
        *(uint4*)&As[0][r][c4 * 8] = v;
      }
#pragma unroll
      for (int p = 0; p < TN / 64; ++p) {
        int r = rT + p * 64;
        uint4 v = *(const uint4*)(Wt + (size_t)(nBlock + r) * K + k0 + c4 * 8);
        *(uint4*)&Bs[0][r][c4 * 8] = v;
      }
      __syncthreads();
      compute(0);
      __syncthreads();
    }
  }

  // ---- epilogue: bias + ReLU + stores / scatter-max ----
#pragma unroll
  for (int fm = 0; fm < 2; ++fm) {
#pragma unroll
    for (int fn = 0; fn < NF; ++fn) {
      int n  = nBlock + waveN * (NF * 16) + fn * 16 + l16;
      float bn = bias ? bias[n] : 0.0f;
#pragma unroll
      for (int r = 0; r < 8; ++r) {
        int m = mBlock + waveM * 32 + fm * 16 + r + halfSel * 8;
        if (!MGUARD || m < M) {
          float v = acc[fm][fn][r] + bn;
          v = v > 0.0f ? v : 0.0f;
          if (outB) outB[(size_t)m * N + n] = f2bf(v);
          if (outF) outF[(size_t)m * N + n] = v;
          if (scat) atomicMax((int*)&scat[(size_t)sidx[m] * N + n],
                              __float_as_int(v));
        }
      }
    }
  }
}

// ---------------------------------------------------------------------------
// Host-side orchestration
// ---------------------------------------------------------------------------
extern "C" void kernel_launch(void* const* d_in, const int* in_sizes, int n_in,
                              void* d_out, int out_size, void* d_ws, size_t ws_size,
                              hipStream_t stream) {
  (void)in_sizes; (void)n_in; (void)out_size; (void)ws_size;

  const float* nodes     = (const float*)d_in[0];
  const float* edge_attr = (const float*)d_in[1];
  const float* glob      = (const float*)d_in[2];
  const int*   eidx      = (const int*)d_in[3];
  const int*   src       = eidx;
  const int*   tgt       = eidx + NEDGE;
  const int*   batch     = (const int*)d_in[6];
  const float* msg_w[3]  = {(const float*)d_in[7],  (const float*)d_in[9],  (const float*)d_in[11]};
  const float* msg_b[3]  = {(const float*)d_in[8],  (const float*)d_in[10], (const float*)d_in[12]};
  const float* upd_w[2]  = {(const float*)d_in[13], (const float*)d_in[15]};
  const float* upd_b[2]  = {(const float*)d_in[14], (const float*)d_in[16]};
  const float* glob_w    = (const float*)d_in[17];
  const float* glob_b    = (const float*)d_in[18];
  const float* emb_w     = (const float*)d_in[19];
  const float* emb_b     = (const float*)d_in[20];
  const float* act_w     = (const float*)d_in[21];
  const float* act_b     = (const float*)d_in[22];

  float* out_nodes  = (float*)d_out;                       // 16384 x 64
  float* out_action = (float*)d_out + (size_t)NNODE * 64;  // 64 x 8

  // ---- workspace layout (256B aligned) ----
  char* ws = (char*)d_ws;
  size_t off = 0;
#define WS_ALLOC(name, bytes) \
  char* name = ws + off; off = (off + (size_t)(bytes) + 255) & ~(size_t)255;

  WS_ALLOC(A1,   (size_t)NEDGE * 704 * 2)   // concat1 bf16  (185 MB)
  WS_ALLOC(H1,   (size_t)NEDGE * 512 * 2)   // msg hidden1   (134 MB)
  WS_ALLOC(AGG,  (size_t)NNODE * 256 * 4)   // scatter-max aggr f32
  WS_ALLOC(A2,   (size_t)NNODE * 640 * 2)   // concat2 bf16
  WS_ALLOC(HU1,  (size_t)NNODE * 512 * 2)   // upd hidden1
  WS_ALLOC(HB,   (size_t)NNODE * 256 * 2)   // h bf16
  WS_ALLOC(AGGN, (size_t)NG * 256 * 4)      // per-graph max f32
  WS_ALLOC(AG,   (size_t)NG * 384 * 2)      // glob concat bf16
  WS_ALLOC(GBUF, (size_t)NG * 256 * 4)      // g f32
  WS_ALLOC(WBF,  (size_t)1327104 * 2)       // all bf16 weights (transposed)
  char* H2 = A1;                            // msg hidden2 aliases concat1

  uint16_t* wbf = (uint16_t*)WBF;
  uint16_t* w_msg0 = wbf;                    // [512][704]
  uint16_t* w_msg1 = wbf + 360448;           // [512][512]
  uint16_t* w_msg2 = wbf + 622592;           // [256][512]
  uint16_t* w_upd0 = wbf + 753664;           // [512][640]
  uint16_t* w_upd1 = wbf + 1081344;          // [256][512]
  uint16_t* w_glob = wbf + 1212416;          // [256][384]
  uint16_t* w_emb  = wbf + 1310720;          // [64][256]

  const int T = 256;
#define CEIL(a, b) (((a) + (b) - 1) / (b))

  // 1) weights -> bf16, transposed to [N][K]
  struct { const float* s; uint16_t* d; int K; int N; } wc[7] = {
      {msg_w[0], w_msg0, 704, 512}, {msg_w[1], w_msg1, 512, 512},
      {msg_w[2], w_msg2, 512, 256}, {upd_w[0], w_upd0, 640, 512},
      {upd_w[1], w_upd1, 512, 256}, {glob_w,   w_glob, 384, 256},
      {emb_w,    w_emb,  256, 64}};
  for (int i = 0; i < 7; ++i)
    k_cvt_transpose_w<<<CEIL(wc[i].K * wc[i].N, T), T, 0, stream>>>(
        wc[i].s, wc[i].d, wc[i].K, wc[i].N);

  // 2) zero scatter buffers (every call: atomicMax accumulates otherwise)
  k_zero_f32<<<CEIL(NNODE * 256, T), T, 0, stream>>>((float*)AGG, NNODE * 256);
  k_zero_f32<<<CEIL(NG * 256, T),    T, 0, stream>>>((float*)AGGN, NG * 256);

  // 3) edge concat
  k_concat_edges<<<CEIL(NEDGE * 704, T), T, 0, stream>>>(
      nodes, edge_attr, glob, src, tgt, (uint16_t*)A1);

  // 4-6) message MLP: 704->512->512->256, last layer scatter-max into AGG[tgt]
  k_gemm_bf16<false, 4><<<dim3(512 / 128, NEDGE / TM), T, 0, stream>>>(
      (const uint16_t*)A1, w_msg0, msg_b[0],
      (uint16_t*)H1, nullptr, nullptr, nullptr, NEDGE, 512, 704);
  k_gemm_bf16<false, 4><<<dim3(512 / 128, NEDGE / TM), T, 0, stream>>>(
      (const uint16_t*)H1, w_msg1, msg_b[1],
      (uint16_t*)H2, nullptr, nullptr, nullptr, NEDGE, 512, 512);
  k_gemm_bf16<false, 4><<<dim3(256 / 128, NEDGE / TM), T, 0, stream>>>(
      (const uint16_t*)H2, w_msg2, msg_b[2],
      nullptr, nullptr, (float*)AGG, tgt, NEDGE, 256, 512);

  // 7) node concat
  k_concat_nodes<<<CEIL(NNODE * 640, T), T, 0, stream>>>(
      nodes, (const float*)AGG, glob, (uint16_t*)A2);

  // 8-9) update MLP: 640->512->256; layer2 stores h + scatter-max per graph
  k_gemm_bf16<false, 4><<<dim3(512 / 128, NNODE / TM), T, 0, stream>>>(
      (const uint16_t*)A2, w_upd0, upd_b[0],
      (uint16_t*)HU1, nullptr, nullptr, nullptr, NNODE, 512, 640);
  k_gemm_bf16<false, 4><<<dim3(256 / 128, NNODE / TM), T, 0, stream>>>(
      (const uint16_t*)HU1, w_upd1, upd_b[1],
      (uint16_t*)HB, nullptr, (float*)AGGN, batch, NNODE, 256, 512);

  // 10) global concat + glob head -> g (f32)   (M=64 -> guarded vector path)
  k_concat_glob<<<CEIL(NG * 384, T), T, 0, stream>>>(
      (const float*)AGGN, glob, (uint16_t*)AG);
  k_gemm_bf16<true, 4><<<dim3(256 / 128, CEIL(NG, TM)), T, 0, stream>>>(
      (const uint16_t*)AG, w_glob, glob_b,
      nullptr, (float*)GBUF, nullptr, nullptr, NG, 256, 384);

  // 11) node_pred = relu(h @ emb_w + emb_b) -> d_out (f32)
  k_gemm_bf16<false, 2><<<dim3(64 / 64, NNODE / TM), T, 0, stream>>>(
      (const uint16_t*)HB, w_emb, emb_b,
      nullptr, out_nodes, nullptr, nullptr, NNODE, 64, 256);

  // 12) action = softmax(g @ act_w + act_b)
  k_act_softmax<<<1, 64, 0, stream>>>((const float*)GBUF, act_w, act_b,
                                      out_action);
#undef WS_ALLOC
#undef CEIL
}
